// LocalMerge_75136157876251
// MI455X (gfx1250) — matrix operand order repaired and checked by reading, not compiled, not run on hardware
//
#include <hip/hip_runtime.h>
#include <float.h>
#include <math.h>

// ---- problem constants (match reference setup_inputs) ----
#define KNNK   32
#define BSZ    4
#define NPTS   2048
#define CIN    128
#define COUT   256
#define BN     (BSZ * NPTS)   // 8192 total points

typedef __attribute__((ext_vector_type(16))) __bf16 v16bf;
typedef __attribute__((ext_vector_type(8)))  __bf16 v8bf;
typedef __attribute__((ext_vector_type(8)))  float  v8f;
typedef __attribute__((ext_vector_type(2)))  float  v2f;

// ---------------------------------------------------------------------------
// f32 -> bf16 conversion (plain, for activation matrices)
// ---------------------------------------------------------------------------
__global__ void cvt_bf16_kernel(const float* __restrict__ s, __bf16* __restrict__ d, int n) {
    int i = blockIdx.x * 256 + threadIdx.x;
    if (i < n) d[i] = (__bf16)s[i];
}

// ---------------------------------------------------------------------------
// Pack W[K][N] (f32) into per-lane bf16 B-fragment layout:
//   dst[(((nt*KC + kc)*32 + lane)*16 + e]  with  K = kc*32 + (lane>=16?16:0) + e,
//   N = nt*16 + (lane&15), KC = K/32.
// Each lane of the GEMM then loads its 16 bf16 B elements as one contiguous
// 32-byte vector instead of 16 strided u16 loads.
// ---------------------------------------------------------------------------
__global__ void pack_w_kernel(const float* __restrict__ W, __bf16* __restrict__ dst,
                              int K, int N) {
    int i = blockIdx.x * 256 + threadIdx.x;
    if (i >= K * N) return;
    int e    = i & 15;
    int lane = (i >> 4) & 31;
    int f    = i >> 9;               // fragment id = nt*KC + kc
    int KC   = K >> 5;
    int kc   = f % KC, nt = f / KC;
    int k = kc * 32 + ((lane >= 16) ? 16 : 0) + e;
    int n = nt * 16 + (lane & 15);
    dst[i] = (__bf16)W[(size_t)k * N + n];
}

// ---------------------------------------------------------------------------
// per-row squared norms of feature [BN, CIN]; one wave32 per row
// ---------------------------------------------------------------------------
__global__ void row_norms_kernel(const float* __restrict__ f, float* __restrict__ out) {
    int w    = (blockIdx.x * blockDim.x + threadIdx.x) >> 5;
    int lane = threadIdx.x & 31;
    if (w >= BN) return;
    const float* r = f + (size_t)w * CIN;
    float s = 0.f;
    for (int c = lane; c < CIN; c += 32) { float v = r[c]; s += v * v; }
    for (int off = 16; off; off >>= 1) s += __shfl_xor(s, off, 32);
    if (lane == 0) out[w] = s;
}

// ---------------------------------------------------------------------------
// KNN in 3-D: queries = xyz, points = base_xyz. One wave per query.
// ---------------------------------------------------------------------------
__global__ void __launch_bounds__(256) knn_xyz_kernel(const float* __restrict__ xyz,
                                                      const float* __restrict__ base,
                                                      int* __restrict__ idx_out) {
    __shared__ float dists[8][NPTS];                 // 64 KB
    int w = threadIdx.x >> 5, lane = threadIdx.x & 31;
    int gq = blockIdx.x * 8 + w;                     // global query id in [0, BN)
    int b  = gq / NPTS;
    const float* q = xyz + (size_t)gq * 3;
    float q0 = q[0], q1 = q[1], q2 = q[2];
    const float* bp = base + (size_t)b * NPTS * 3;
    float* sl = dists[w];
    for (int j = lane; j < NPTS; j += 32) {
        float d0 = q0 - bp[j * 3 + 0];
        float d1 = q1 - bp[j * 3 + 1];
        float d2 = q2 - bp[j * 3 + 2];
        sl[j] = d0 * d0 + d1 * d1 + d2 * d2;
    }
    __syncthreads();
    int* orow = idx_out + (size_t)gq * KNNK;
    for (int p = 0; p < KNNK; ++p) {
        float best = __builtin_inff(); int bj = 0x7fffffff;
        for (int j = lane; j < NPTS; j += 32) {
            float d = sl[j];
            if (d < best || (d == best && j < bj)) { best = d; bj = j; }
        }
        for (int off = 16; off; off >>= 1) {
            float od = __shfl_xor(best, off, 32);
            int   oj = __shfl_xor(bj,   off, 32);
            if (od < best || (od == best && oj < bj)) { best = od; bj = oj; }
        }
        if (lane == 0) { orow[p] = bj; sl[bj] = __builtin_inff(); }
        __syncthreads();
    }
}

// ---------------------------------------------------------------------------
// Feature-space KNN, fused. f32 WMMA (16x16x4) Gram tile, A fragments cached
// in registers for the whole block row (reused across all 16 column tiles),
// B fragments batch-loaded per tile so the loads pipeline ahead of the WMMA
// chain. Scores land in LDS (16 x 2048 f32 = 128 KB), then in-LDS top-32.
// Ranking key: n_j - 2*dot (same order as squared distance; n_i constant/row).
// ---------------------------------------------------------------------------
__global__ void __launch_bounds__(256) knn_feat_kernel(const float* __restrict__ feat,
                                                       const float* __restrict__ norms,
                                                       int* __restrict__ idx_out) {
    extern __shared__ float smem[];
    float* S = smem;                    // [16][NPTS]
    int tid = threadIdx.x, lane = tid & 31, w = tid >> 5;
    int m0 = blockIdx.x * 16;           // global row base (rows stay in one batch)
    int b  = m0 / NPTS;
    const float* fb = feat + (size_t)b * NPTS * CIN;

    int mloc = lane & 15;
    int ksel = (lane >> 4) * 2;         // lanes 16-31 hold K+2, K+3 halves
    int ncol = lane & 15;

    // A fragments: row m0+mloc, all K. 32 v2f pairs, register-resident.
    const float* arow = feat + (size_t)(m0 + mloc) * CIN;
    v2f areg[CIN / 4];
    #pragma unroll
    for (int kb = 0; kb < CIN / 4; ++kb)
        areg[kb] = *(const v2f*)(arow + kb * 4 + ksel);

    for (int ct = w; ct < NPTS / 16; ct += 8) {
        int j0 = ct * 16;
        const float* fcol = fb + (size_t)(j0 + ncol) * CIN;
        if (ct + 8 < NPTS / 16)
            __builtin_prefetch(fcol + (size_t)128 * CIN, 0, 1);   // next tile for this wave
        // batch-load all B fragments for this column tile
        v2f breg[CIN / 4];
        #pragma unroll
        for (int kb = 0; kb < CIN / 4; ++kb)
            breg[kb] = *(const v2f*)(fcol + kb * 4 + ksel);
        v8f acc = {};
#if __has_builtin(__builtin_amdgcn_wmma_f32_16x16x4_f32)
        #pragma unroll
        for (int kb = 0; kb < CIN / 4; ++kb)
            acc = __builtin_amdgcn_wmma_f32_16x16x4_f32(false, areg[kb], false, breg[kb],
                                                        (short)0, acc, false, false);
#else
        #pragma unroll
        for (int kb = 0; kb < CIN / 4; ++kb) {
            // fallback: per-lane partial MACs consistent with C/D layout
            #pragma unroll
            for (int g = 0; g < 8; ++g) {
                int r = g + ((lane >= 16) ? 8 : 0);
                const float* ar = feat + (size_t)(m0 + r) * CIN;
                acc[g] += ar[kb * 4 + ksel] * breg[kb][0] + ar[kb * 4 + ksel + 1] * breg[kb][1];
            }
        }
#endif
        float nj = norms[(size_t)b * NPTS + j0 + ncol];
        #pragma unroll
        for (int g = 0; g < 8; ++g) {
            int r = g + ((lane >= 16) ? 8 : 0);   // C/D layout: lanes 16-31 -> M=g+8
            S[r * NPTS + j0 + ncol] = nj - 2.0f * acc[g];
        }
    }
    __syncthreads();

    // top-32 per row; wave w owns rows 2w, 2w+1
    for (int rr = 0; rr < 2; ++rr) {
        int r = w * 2 + rr;
        float* Sr = S + r * NPTS;
        int* orow = idx_out + (size_t)(m0 + r) * KNNK;
        for (int p = 0; p < KNNK; ++p) {
            float best = __builtin_inff(); int bj = 0x7fffffff;
            for (int j = lane; j < NPTS; j += 32) {
                float d = Sr[j];
                if (d < best || (d == best && j < bj)) { best = d; bj = j; }
            }
            for (int off = 16; off; off >>= 1) {
                float od = __shfl_xor(best, off, 32);
                int   oj = __shfl_xor(bj,   off, 32);
                if (od < best || (od == best && oj < bj)) { best = od; bj = oj; }
            }
            if (lane == 0) { orow[p] = bj; Sr[bj] = __builtin_inff(); }
            __builtin_amdgcn_wave_barrier();  // LDS per-wave in-order (DScnt)
        }
    }
}

// ---------------------------------------------------------------------------
// bf16 WMMA GEMM: C[M,N] = A[M,K] * Wp + bias, Wp pre-packed per-lane
// (see pack_w_kernel). One wave per 16x16 output tile, K chunks of 32.
// A fragment: two contiguous 16-byte loads; B fragment: one 32-byte load.
// grid = (M/16, N/128), block = 256 (8 waves -> 8 adjacent N tiles).
// ---------------------------------------------------------------------------
__global__ void __launch_bounds__(256) gemm_bf16_kernel(const __bf16* __restrict__ A,
                                                        const __bf16* __restrict__ Wp,
                                                        const float* __restrict__ bias,
                                                        float* __restrict__ C,
                                                        int M, int Kdim, int Nm) {
    int lane = threadIdx.x & 31, w = threadIdx.x >> 5;
    int mtile = blockIdx.x;
    int ntile = blockIdx.y * 8 + w;
    int m   = mtile * 16 + (lane & 15);
    int col = ntile * 16 + (lane & 15);
    int khA = (lane >= 16) ? 16 : 0;    // A: lane-half selects contiguous K half of 16
    int KC  = Kdim >> 5;
    v8f acc = {};
    for (int kc = 0; kc < Kdim; kc += 32) {
#if __has_builtin(__builtin_amdgcn_wmma_f32_16x16x32_bf16)
        const __bf16* ap = A + (size_t)m * Kdim + kc;
        v8bf alo = *(const v8bf*)(ap + khA);          // K = kc+khA .. +7
        v8bf ahi = *(const v8bf*)(ap + khA + 8);      // K = kc+khA+8 .. +15
        v16bf a = __builtin_shufflevector(alo, ahi, 0, 1, 2, 3, 4, 5, 6, 7,
                                          8, 9, 10, 11, 12, 13, 14, 15);
        const __bf16* wp = Wp + (((size_t)ntile * KC + (kc >> 5)) * 32 + lane) * 16;
        v16bf bb = *(const v16bf*)wp;
        if (kc + 32 < Kdim) {
            __builtin_prefetch(ap + 32, 0, 1);
            __builtin_prefetch(wp + 32 * 16, 0, 1);
        }
        acc = __builtin_amdgcn_wmma_f32_16x16x32_bf16(false, a, false, bb,
                                                      (short)0, acc, false, false);
#else
        int rb = mtile * 16 + ((lane >= 16) ? 8 : 0);
        #pragma unroll
        for (int g = 0; g < 8; ++g)
            for (int k = kc; k < kc + 32; ++k) {
                int within = k & 31;
                int bl = ((within >= 16) ? 16 : 0) + (col & 15);
                int be = within & 15;
                float wv = (float)Wp[(((size_t)ntile * KC + (k >> 5)) * 32 + bl) * 16 + be];
                acc[g] += (float)A[(size_t)(rb + g) * Kdim + k] * wv;
            }
#endif
    }
    float bv = bias ? bias[col] : 0.f;
    int rbase = mtile * 16 + ((lane >= 16) ? 8 : 0);
    #pragma unroll
    for (int g = 0; g < 8; ++g)
        C[(size_t)(rbase + g) * Nm + col] = acc[g] + bv;
}

// ---------------------------------------------------------------------------
// BatchNorm training-mode stats: biased mean/var over M rows per channel.
// ---------------------------------------------------------------------------
__global__ void chan_stats_kernel(const float* __restrict__ X, int M, int Cdim,
                                  float* __restrict__ mean, float* __restrict__ var) {
    __shared__ float s1[256], s2[256];
    int c = blockIdx.x, t = threadIdx.x;
    float a = 0.f, b = 0.f;
    for (int r = t; r < M; r += 256) { float v = X[(size_t)r * Cdim + c]; a += v; b += v * v; }
    s1[t] = a; s2[t] = b; __syncthreads();
    for (int off = 128; off; off >>= 1) {
        if (t < off) { s1[t] += s1[t + off]; s2[t] += s2[t + off]; }
        __syncthreads();
    }
    if (t == 0) { float mu = s1[0] / M; mean[c] = mu; var[c] = s2[0] / M - mu * mu; }
}

// ---------------------------------------------------------------------------
// y = leaky(gamma*(x-mean)*rsqrt(var+1e-5)+beta, 0.2) [+ add]; placed at
// Y[r*ostride + ocol + c] so branch outputs concat directly.
// ---------------------------------------------------------------------------
__global__ void bn_leaky_kernel(const float* __restrict__ X, const float* __restrict__ mean,
                                const float* __restrict__ var, const float* __restrict__ gamma,
                                const float* __restrict__ beta, const float* __restrict__ add,
                                float* __restrict__ Y, int M, int Cdim, int ostride, int ocol) {
    int i = blockIdx.x * 256 + threadIdx.x;
    if (i >= M * Cdim) return;
    int r = i / Cdim, c = i % Cdim;
    float y = gamma[c] * (X[i] - mean[c]) * rsqrtf(var[c] + 1e-5f) + beta[c];
    y = (y >= 0.f) ? y : 0.2f * y;
    if (add) y += add[i];
    Y[(size_t)r * ostride + ocol + c] = y;
}

// ---------------------------------------------------------------------------
// Attention: per (point, channel): e_k=(q-k_gather)/16; softmax over K;
// att-1; ctx = max_k (att-1)*v_gather. One block (COUT threads) per point.
// ---------------------------------------------------------------------------
__global__ void __launch_bounds__(256) attention_kernel(const float* __restrict__ Q,
                                                        const float* __restrict__ Km,
                                                        const float* __restrict__ V,
                                                        const int* __restrict__ idx,
                                                        float* __restrict__ ctx) {
    __shared__ int js[KNNK];
    int row = blockIdx.x;                 // [0, BN)
    int b   = row / NPTS;
    int c   = threadIdx.x;
    if (c < KNNK) js[c] = idx[(size_t)row * KNNK + c];
    __syncthreads();
    size_t base = (size_t)b * NPTS * COUT;
    float qv = Q[(size_t)row * COUT + c];
    float e[KNNK];
    float mx = -__builtin_inff();
    #pragma unroll
    for (int k = 0; k < KNNK; ++k) {
        float kv = Km[base + (size_t)js[k] * COUT + c];
        e[k] = (qv - kv) * 0.0625f;       // 1/sqrt(256)
        mx = fmaxf(mx, e[k]);
    }
    float s = 0.f;
    #pragma unroll
    for (int k = 0; k < KNNK; ++k) { e[k] = __expf(e[k] - mx); s += e[k]; }
    float inv = 1.f / s;
    float m = -__builtin_inff();
    #pragma unroll
    for (int k = 0; k < KNNK; ++k) {
        float att = e[k] * inv - 1.0f;    // softmax - sum(softmax) == att - 1
        float vv  = V[base + (size_t)js[k] * COUT + c];
        m = fmaxf(m, att * vv);
    }
    ctx[(size_t)row * COUT + c] = m;
}

// ---------------------------------------------------------------------------
extern "C" void kernel_launch(void* const* d_in, const int* in_sizes, int n_in,
                              void* d_out, int out_size, void* d_ws, size_t ws_size,
                              hipStream_t stream) {
    (void)in_sizes; (void)n_in; (void)out_size; (void)ws_size;
    const float* xyz       = (const float*)d_in[0];
    const float* base_xyz  = (const float*)d_in[1];
    const float* feature   = (const float*)d_in[2];
    const float* qW  = (const float*)d_in[3];  const float* qb  = (const float*)d_in[4];
    const float* kW  = (const float*)d_in[5];  const float* kb  = (const float*)d_in[6];
    const float* vW  = (const float*)d_in[7];  const float* vb  = (const float*)d_in[8];
    const float* resW = (const float*)d_in[9]; const float* resb = (const float*)d_in[10];
    const float* res_gamma = (const float*)d_in[11]; const float* res_beta = (const float*)d_in[12];
    const float* ffnW = (const float*)d_in[13]; const float* ffnb = (const float*)d_in[14];
    const float* ffn_gamma = (const float*)d_in[15]; const float* ffn_beta = (const float*)d_in[16];
    const float* fcW = (const float*)d_in[17]; const float* fcb = (const float*)d_in[18];
    const float* fc_gamma = (const float*)d_in[19]; const float* fc_beta = (const float*)d_in[20];

    char* ws = (char*)d_ws;
    size_t off = 0;
    auto take = [&](size_t bytes) -> void* {
        void* p = ws + off;
        off = (off + bytes + 255) & ~(size_t)255;
        return p;
    };
    __bf16* feat_bf   = (__bf16*)take((size_t)BN * CIN * 2);
    __bf16* qw_bf     = (__bf16*)take((size_t)2 * CIN * COUT * 2);
    __bf16* kw_bf     = (__bf16*)take((size_t)2 * CIN * COUT * 2);
    __bf16* vw_bf     = (__bf16*)take((size_t)2 * CIN * COUT * 2);
    __bf16* rw_bf     = (__bf16*)take((size_t)2 * CIN * COUT * 2);
    __bf16* ffnw_bf   = (__bf16*)take((size_t)2 * COUT * COUT * 2);
    __bf16* fcw_bf    = (__bf16*)take((size_t)2 * COUT * COUT * 2);   // 512x256
    float*  norms     = (float*)take((size_t)BN * 4);
    int*    idx_xyz   = (int*)take((size_t)BN * KNNK * 4);
    int*    idx_feat  = (int*)take((size_t)BN * KNNK * 4);
    float*  bufQ      = (float*)take((size_t)BN * COUT * 4);
    float*  bufK      = (float*)take((size_t)BN * COUT * 4);
    float*  bufV      = (float*)take((size_t)BN * COUT * 4);
    float*  bufR      = (float*)take((size_t)BN * COUT * 4);
    float*  ctx       = (float*)take((size_t)BN * COUT * 4);
    __bf16* ctx_bf    = (__bf16*)take((size_t)BN * COUT * 2);
    float*  ffnout    = (float*)take((size_t)BN * COUT * 4);
    float*  concatb   = (float*)take((size_t)BN * 2 * COUT * 4);
    __bf16* concat_bf = (__bf16*)take((size_t)BN * 2 * COUT * 2);
    float*  fcout     = (float*)take((size_t)BN * COUT * 4);
    float*  meanb     = (float*)take((size_t)COUT * 4);
    float*  varb      = (float*)take((size_t)COUT * 4);

    // --- activations -> bf16; weights -> packed bf16 B-fragments ---
    cvt_bf16_kernel<<<(BN * CIN + 255) / 256, 256, 0, stream>>>(feature, feat_bf, BN * CIN);
    const int WG1 = (CIN * COUT + 255) / 256;    // per-branch 128x256 pack
    const int WG2 = (COUT * COUT + 255) / 256;   // per-branch 256x256 pack
    for (int t = 0; t < 2; ++t) {
        pack_w_kernel<<<WG1, 256, 0, stream>>>(qW + (size_t)t * CIN * COUT,
                                               qw_bf + (size_t)t * CIN * COUT, CIN, COUT);
        pack_w_kernel<<<WG1, 256, 0, stream>>>(kW + (size_t)t * CIN * COUT,
                                               kw_bf + (size_t)t * CIN * COUT, CIN, COUT);
        pack_w_kernel<<<WG1, 256, 0, stream>>>(vW + (size_t)t * CIN * COUT,
                                               vw_bf + (size_t)t * CIN * COUT, CIN, COUT);
        pack_w_kernel<<<WG1, 256, 0, stream>>>(resW + (size_t)t * CIN * COUT,
                                               rw_bf + (size_t)t * CIN * COUT, CIN, COUT);
        pack_w_kernel<<<WG2, 256, 0, stream>>>(ffnW + (size_t)t * COUT * COUT,
                                               ffnw_bf + (size_t)t * COUT * COUT, COUT, COUT);
    }
    pack_w_kernel<<<(2 * COUT * COUT + 255) / 256, 256, 0, stream>>>(fcW, fcw_bf, 2 * COUT, COUT);

    // --- KNNs ---
    row_norms_kernel<<<(BN * 32) / 256, 256, 0, stream>>>(feature, norms);
    knn_xyz_kernel<<<BN / 8, 256, 0, stream>>>(xyz, base_xyz, idx_xyz);
    size_t smem_feat = (size_t)(16 * NPTS) * sizeof(float);   // 128 KB < 320 KB/WGP
    knn_feat_kernel<<<BN / 16, 256, smem_feat, stream>>>(feature, norms, idx_feat);

    // --- two LocalTrans branches ---
    dim3 ggrid(BN / 16, COUT / 128);
    for (int t = 0; t < 2; ++t) {
        const __bf16* wq = qw_bf + (size_t)t * CIN * COUT;
        const __bf16* wk = kw_bf + (size_t)t * CIN * COUT;
        const __bf16* wv = vw_bf + (size_t)t * CIN * COUT;
        const __bf16* wr = rw_bf + (size_t)t * CIN * COUT;
        gemm_bf16_kernel<<<ggrid, 256, 0, stream>>>(feat_bf, wq, qb + t * COUT, bufQ, BN, CIN, COUT);
        gemm_bf16_kernel<<<ggrid, 256, 0, stream>>>(feat_bf, wk, kb + t * COUT, bufK, BN, CIN, COUT);
        gemm_bf16_kernel<<<ggrid, 256, 0, stream>>>(feat_bf, wv, vb + t * COUT, bufV, BN, CIN, COUT);
        gemm_bf16_kernel<<<ggrid, 256, 0, stream>>>(feat_bf, wr, resb + t * COUT, bufR, BN, CIN, COUT);

        // residual = leaky(BN(R)) in place
        chan_stats_kernel<<<COUT, 256, 0, stream>>>(bufR, BN, COUT, meanb, varb);
        bn_leaky_kernel<<<(BN * COUT) / 256, 256, 0, stream>>>(bufR, meanb, varb,
            res_gamma + t * COUT, res_beta + t * COUT, nullptr, bufR, BN, COUT, COUT, 0);

        attention_kernel<<<BN, 256, 0, stream>>>(bufQ, bufK, bufV,
            (t == 0) ? idx_xyz : idx_feat, ctx);

        // ffn: leaky(BN(ctx @ ffnW + ffnb)) + residual -> concat slot t
        cvt_bf16_kernel<<<(BN * COUT + 255) / 256, 256, 0, stream>>>(ctx, ctx_bf, BN * COUT);
        gemm_bf16_kernel<<<ggrid, 256, 0, stream>>>(ctx_bf, ffnw_bf + (size_t)t * COUT * COUT,
            ffnb + t * COUT, ffnout, BN, COUT, COUT);
        chan_stats_kernel<<<COUT, 256, 0, stream>>>(ffnout, BN, COUT, meanb, varb);
        bn_leaky_kernel<<<(BN * COUT) / 256, 256, 0, stream>>>(ffnout, meanb, varb,
            ffn_gamma + t * COUT, ffn_beta + t * COUT, bufR, concatb, BN, COUT, 2 * COUT, t * COUT);
    }

    // --- fusion MLP: leaky(BN(concat @ fcW + fcb)) ---
    cvt_bf16_kernel<<<(BN * 2 * COUT + 255) / 256, 256, 0, stream>>>(concatb, concat_bf, BN * 2 * COUT);
    gemm_bf16_kernel<<<ggrid, 256, 0, stream>>>(concat_bf, fcw_bf, fcb, fcout, BN, 2 * COUT, COUT);
    chan_stats_kernel<<<COUT, 256, 0, stream>>>(fcout, BN, COUT, meanb, varb);
    bn_leaky_kernel<<<(BN * COUT) / 256, 256, 0, stream>>>(fcout, meanb, varb,
        fc_gamma, fc_beta, nullptr, (float*)d_out, BN, COUT, COUT, 0);
}
